// LSTMNet_42460046688746
// MI455X (gfx1250) — matrix operand order repaired and checked by reading, not compiled, not run on hardware
//
#include <hip/hip_runtime.h>
#include <math.h>

// ---------------------------------------------------------------------------
// 2-layer LSTM for MI455X (gfx1250, wave32).
//   zx = x@W + b  : big bf16 WMMA GEMMs (parallel, compute bound)
//   scan          : persistent 8-WGP kernel, per-step 32x512x2048 bf16 WMMA
//                   GEMM + LDS gate exchange + device-wide spin barrier.
//                   Next-step zx is register-double-buffered across the
//                   barrier to hide HBM latency; t+2 is prefetched to cache.
// ---------------------------------------------------------------------------

typedef __attribute__((ext_vector_type(16))) __bf16 v16bf;
typedef __attribute__((ext_vector_type(8)))  __bf16 v8bf;
typedef __attribute__((ext_vector_type(8)))  float  v8f;

#define B_  32
#define T_  1024
#define DIN 256
#define H_  512
#define G4  2048   // 4*H
#define NSCAN_WG 8

// ---- WMMA 16x16x32 bf16 fragment loaders (CDNA5 VGPR layouts) -------------
// A (MxK, 16-bit): lane l -> row (l&15); lanes 0-15 hold K {0..7,16..23},
// lanes 16-31 hold K {8..15,24..31}  => two b128 loads per lane.
__device__ __forceinline__ v16bf load_frag_a(const __bf16* __restrict__ base,
                                             int ld, int row0, int k0) {
  const int l = threadIdx.x & 31;
  const int m = row0 + (l & 15);
  const int grp = l >> 4;
  const __bf16* p = base + (size_t)m * ld + k0 + grp * 8;
  v8bf lo = *reinterpret_cast<const v8bf*>(p);
  v8bf hi = *reinterpret_cast<const v8bf*>(p + 16);
  v16bf r;
#pragma unroll
  for (int i = 0; i < 8; ++i) { r[i] = lo[i]; r[i + 8] = hi[i]; }
  return r;
}

// B (KxN, 16-bit): lane l -> col (l&15); lanes 0-15 hold K {0..7,16..23},
// lanes 16-31 hold K {8..15,24..31} (same as A). Bt is stored N-major (ld = K),
// so this is two b128 loads per lane.
__device__ __forceinline__ v16bf load_frag_b(const __bf16* __restrict__ bt,
                                             int ld, int col0, int k0) {
  const int l = threadIdx.x & 31;
  const int n = col0 + (l & 15);
  const int grp = l >> 4;
  const __bf16* p = bt + (size_t)n * ld + k0 + grp * 8;
  v8bf lo = *reinterpret_cast<const v8bf*>(p);
  v8bf hi = *reinterpret_cast<const v8bf*>(p + 16);
  v16bf r;
#pragma unroll
  for (int i = 0; i < 8; ++i) { r[i] = lo[i]; r[i + 8] = hi[i]; }
  return r;
}

#define WMMA_BF16(A, Bv, C) \
  __builtin_amdgcn_wmma_f32_16x16x32_bf16(false, (A), false, (Bv), (short)0, (C), false, false)

__device__ __forceinline__ float sigmoidf_(float x) {
  return 1.0f / (1.0f + __expf(-x));
}
__device__ __forceinline__ float tanhf_(float x) {
  float t = __expf(-2.0f * fabsf(x));
  float r = (1.0f - t) / (1.0f + t);
  return copysignf(r, x);
}

// ---- conversion helpers ----------------------------------------------------
__global__ void cvt_bf16(const float* __restrict__ in, __bf16* __restrict__ out, int n) {
  int i = blockIdx.x * 256 + threadIdx.x;
  if (i < n) out[i] = (__bf16)in[i];
}

// out[n*K + k] = (bf16) in[k*N + n]   (N-major transpose for B operands)
__global__ void transpose_cvt(const float* __restrict__ in, __bf16* __restrict__ out,
                              int K, int N) {
  int i = blockIdx.x * 256 + threadIdx.x;
  if (i < K * N) {
    int n = i / K, k = i - n * K;
    out[i] = (__bf16)in[(size_t)k * N + n];
  }
}

// ---- input-projection GEMM: C[M,2048] = A[M,K] @ Bt[2048,K]^T + bias ------
// 256 threads = 8 waves (2x4); each wave computes a 32x64 tile (2x4 WMMA).
__global__ __launch_bounds__(256) void gemm_bias_bf16(
    const __bf16* __restrict__ A, const __bf16* __restrict__ Bt,
    const float* __restrict__ bias, float* __restrict__ C, int K) {
  const int w  = threadIdx.x >> 5;
  const int wr = w >> 2;                  // 0..1
  const int wc = w & 3;                   // 0..3
  const int row0 = blockIdx.y * 64 + wr * 32;
  const int col0 = blockIdx.x * 256 + wc * 64;
  const int l = threadIdx.x & 31;
  const int grp = l >> 4;

  v8f acc[2][4];
#pragma unroll
  for (int mt = 0; mt < 2; ++mt)
#pragma unroll
    for (int nt = 0; nt < 4; ++nt)
      acc[mt][nt] = (v8f){0.f, 0.f, 0.f, 0.f, 0.f, 0.f, 0.f, 0.f};

  for (int k0 = 0; k0 < K; k0 += 32) {
    v16bf a0 = load_frag_a(A, K, row0,      k0);
    v16bf a1 = load_frag_a(A, K, row0 + 16, k0);
#pragma unroll
    for (int nt = 0; nt < 4; ++nt) {
      v16bf b = load_frag_b(Bt, K, col0 + nt * 16, k0);
      acc[0][nt] = WMMA_BF16(a0, b, acc[0][nt]);
      acc[1][nt] = WMMA_BF16(a1, b, acc[1][nt]);
    }
  }

#pragma unroll
  for (int mt = 0; mt < 2; ++mt)
#pragma unroll
    for (int nt = 0; nt < 4; ++nt) {
      const int n = col0 + nt * 16 + (l & 15);
      const float bv = bias[n];
#pragma unroll
      for (int r = 0; r < 8; ++r) {
        const int row = row0 + mt * 16 + r + 8 * grp;   // C/D layout: VGPR r -> M = r + 8*grp
        C[(size_t)row * G4 + n] = acc[mt][nt][r] + bv;
      }
    }
}

// ---- persistent recurrent scan --------------------------------------------
// 8 WGs; WG wg owns gate-columns j0 = wg*64 .. +63 for all 4 gates.
// 8 waves: wave w -> gate = w>>1, column-half = w&1. Per step each wave does
// 2x2 WMMA tiles over K=512 (64 WMMAs). Gates exchanged via LDS; h/c state
// in global; device-wide spin barrier per step.
__device__ __forceinline__ void load_zx_step(const float* __restrict__ zx, int t,
                                             int gate, int hh, int j0, int l, int grp,
                                             v8f z[2][2]) {
#pragma unroll
  for (int mt = 0; mt < 2; ++mt)
#pragma unroll
    for (int nt = 0; nt < 2; ++nt) {
      const int nabs = gate * H_ + j0 + hh * 32 + nt * 16 + (l & 15);
#pragma unroll
      for (int r = 0; r < 8; ++r) {
        const int m = mt * 16 + r + 8 * grp;              // batch row
        z[mt][nt][r] = zx[((size_t)m * T_ + t) * G4 + nabs];
      }
    }
}

__global__ __launch_bounds__(256) void lstm_scan(
    const float* __restrict__ zx,   // [B*T, 2048] (row = b*T + t)
    const __bf16* __restrict__ Ut,  // [2048, 512] N-major
    __bf16* __restrict__ hbuf,      // [32, 512] bf16 recurrent state
    float* __restrict__ cbuf,       // [32, 512] f32 cell state
    __bf16* __restrict__ hall_bf,   // [B*T, 512] bf16 (layer-0 output) or null
    float* __restrict__ outf,       // [B*T, 512] f32 (final output) or null
    unsigned int* __restrict__ cnt) {
  __shared__ float zlds[4 * 32 * 64];   // 32 KB: [gate][row 0..31][col 0..63]

  const int tid = threadIdx.x;
  const int w = tid >> 5;
  const int l = tid & 31;
  const int grp = l >> 4;
  const int gate = w >> 1;
  const int hh = w & 1;
  const int j0 = blockIdx.x * 64;

  // Register double buffer of the zx slice for the upcoming step: the loads
  // are issued before the inter-WG barrier so the HBM latency overlaps the
  // barrier spin instead of serializing with the next step's GEMM.
  v8f znext[2][2];
  load_zx_step(zx, 0, gate, hh, j0, l, grp, znext);

  for (int t = 0; t < T_; ++t) {
    // ---- z = zx[:, t, :] as WMMA accumulator init -------------------------
    v8f acc[2][2];
#pragma unroll
    for (int mt = 0; mt < 2; ++mt)
#pragma unroll
      for (int nt = 0; nt < 2; ++nt)
        acc[mt][nt] = znext[mt][nt];

    // ---- z += h_{t-1} @ U  (skipped at t=0 since h=0) ---------------------
    if (t > 0) {
      const int ncol0 = gate * H_ + j0 + hh * 32;
      for (int k0 = 0; k0 < H_; k0 += 32) {
        v16bf a0 = load_frag_a(hbuf, H_, 0,  k0);
        v16bf a1 = load_frag_a(hbuf, H_, 16, k0);
        v16bf b0 = load_frag_b(Ut, H_, ncol0,      k0);
        v16bf b1 = load_frag_b(Ut, H_, ncol0 + 16, k0);
        acc[0][0] = WMMA_BF16(a0, b0, acc[0][0]);
        acc[0][1] = WMMA_BF16(a0, b1, acc[0][1]);
        acc[1][0] = WMMA_BF16(a1, b0, acc[1][0]);
        acc[1][1] = WMMA_BF16(a1, b1, acc[1][1]);
      }
    }

    // ---- publish z tiles to LDS ------------------------------------------
#pragma unroll
    for (int mt = 0; mt < 2; ++mt)
#pragma unroll
      for (int nt = 0; nt < 2; ++nt) {
        const int col = hh * 32 + nt * 16 + (l & 15);
#pragma unroll
        for (int r = 0; r < 8; ++r) {
          const int m = mt * 16 + r + 8 * grp;
          zlds[((gate * 32 + m) << 6) + col] = acc[mt][nt][r];
        }
      }
    __syncthreads();

    // ---- gate math: 32*64 elements / 256 threads = 8 each -----------------
#pragma unroll
    for (int q = 0; q < 8; ++q) {
      const int idx = q * 256 + tid;
      const int m = idx >> 6;           // batch
      const int jl = idx & 63;          // local H column
      const float iv = sigmoidf_(zlds[((0 * 32 + m) << 6) + jl]);
      const float fv = sigmoidf_(zlds[((1 * 32 + m) << 6) + jl]);
      const float gv = tanhf_  (zlds[((2 * 32 + m) << 6) + jl]);
      const float ov = sigmoidf_(zlds[((3 * 32 + m) << 6) + jl]);
      const int cj = m * H_ + j0 + jl;
      const float cp = (t == 0) ? 0.0f : cbuf[cj];
      const float cn = fv * cp + iv * gv;
      cbuf[cj] = cn;
      const float hv = ov * tanhf_(cn);
      hbuf[cj] = (__bf16)hv;
      const size_t oj = ((size_t)m * T_ + t) * H_ + j0 + jl;
      if (hall_bf) hall_bf[oj] = (__bf16)hv;
      if (outf)    outf[oj]    = hv;
    }

    // ---- issue next-step zx loads + t+2 cache prefetch (overlap barrier) --
    if (t + 1 < T_) {
      load_zx_step(zx, t + 1, gate, hh, j0, l, grp, znext);
    }
    if (t + 2 < T_) {
      // lane l covers batch row l; two prefetches cover the wave's 32
      // gate-columns (global_prefetch_b8 pulls whole cachelines).
      const int nab0 = gate * H_ + j0 + hh * 32;
      const float* pf = &zx[((size_t)l * T_ + (t + 2)) * G4 + nab0];
      __builtin_prefetch(pf, 0, 1);
      __builtin_prefetch(pf + 16, 0, 1);
    }

    // ---- device-wide step barrier ----------------------------------------
    __threadfence();
    __syncthreads();
    if (tid == 0) {
      __hip_atomic_fetch_add(cnt, 1u, __ATOMIC_RELAXED, __HIP_MEMORY_SCOPE_AGENT);
      const unsigned target = (unsigned)NSCAN_WG * (unsigned)(t + 1);
      // RELAXED polls: device-scope atomic loads stay coherent without
      // per-poll cache-invalidate traffic; acquire fence follows below.
      while (__hip_atomic_load(cnt, __ATOMIC_RELAXED, __HIP_MEMORY_SCOPE_AGENT) < target) {
        __builtin_amdgcn_s_sleep(2);
      }
    }
    __syncthreads();
    __threadfence();
  }
}

// ---------------------------------------------------------------------------
extern "C" void kernel_launch(void* const* d_in, const int* in_sizes, int n_in,
                              void* d_out, int out_size, void* d_ws, size_t ws_size,
                              hipStream_t stream) {
  const float* x  = (const float*)d_in[0];
  const float* W0 = (const float*)d_in[1];
  const float* U0 = (const float*)d_in[2];
  const float* b0 = (const float*)d_in[3];
  const float* W1 = (const float*)d_in[4];
  const float* U1 = (const float*)d_in[5];
  const float* b1 = (const float*)d_in[6];
  float* out = (float*)d_out;

  char* ws = (char*)d_ws;
  size_t off = 0;
  auto carve = [&](size_t bytes) -> void* {
    void* p = ws + off;
    off += (bytes + 255) & ~(size_t)255;
    return p;
  };
  float*    zx   = (float*)   carve((size_t)B_ * T_ * G4 * sizeof(float));   // 256 MB (reused)
  __bf16*   xbf  = (__bf16*)  carve((size_t)B_ * T_ * DIN * 2);
  __bf16*   h0bf = (__bf16*)  carve((size_t)B_ * T_ * H_ * 2);
  __bf16*   W0t  = (__bf16*)  carve((size_t)G4 * DIN * 2);
  __bf16*   U0t  = (__bf16*)  carve((size_t)G4 * H_ * 2);
  __bf16*   W1t  = (__bf16*)  carve((size_t)G4 * H_ * 2);
  __bf16*   U1t  = (__bf16*)  carve((size_t)G4 * H_ * 2);
  __bf16*   hbuf = (__bf16*)  carve((size_t)B_ * H_ * 2);
  float*    cbuf = (float*)   carve((size_t)B_ * H_ * sizeof(float));
  unsigned* cnt  = (unsigned*)carve(256);

  // Only the barrier counters need re-init per replay (hbuf/cbuf are
  // special-cased at t=0; everything else is written before it is read).
  hipMemsetAsync(cnt, 0, 256, stream);

  cvt_bf16<<<(B_ * T_ * DIN) / 256, 256, 0, stream>>>(x, xbf, B_ * T_ * DIN);
  transpose_cvt<<<(G4 * DIN) / 256, 256, 0, stream>>>(W0, W0t, DIN, G4);
  transpose_cvt<<<(G4 * H_) / 256, 256, 0, stream>>>(U0, U0t, H_, G4);
  transpose_cvt<<<(G4 * H_) / 256, 256, 0, stream>>>(W1, W1t, H_, G4);
  transpose_cvt<<<(G4 * H_) / 256, 256, 0, stream>>>(U1, U1t, H_, G4);

  // Layer 0
  gemm_bias_bf16<<<dim3(G4 / 256, (B_ * T_) / 64), 256, 0, stream>>>(xbf, W0t, b0, zx, DIN);
  lstm_scan<<<NSCAN_WG, 256, 0, stream>>>(zx, U0t, hbuf, cbuf, h0bf, nullptr, cnt);

  // Layer 1
  gemm_bias_bf16<<<dim3(G4 / 256, (B_ * T_) / 64), 256, 0, stream>>>(h0bf, W1t, b1, zx, H_);
  lstm_scan<<<NSCAN_WG, 256, 0, stream>>>(zx, U1t, hbuf, cbuf, nullptr, out, cnt + 32);
}
